// LlavaViTDecoder_77850577207550
// MI455X (gfx1250) — compile-verified, matrix operand order checked by reading, not run on hardware
//
#include <hip/hip_runtime.h>
#include <hip/hip_bf16.h>

// ---------------- fixed problem dims ----------------
#define BB 2
#define LL 2048
#define DD 384
#define NH 6
#define HD 48
#define FF 1536
#define NL 8
#define NVIS 512
#define MTOT (BB*LL)          // 4096 rows
#define QKVN (3*DD)           // 1152
#define HDP 64                // padded head dim

typedef __attribute__((ext_vector_type(16))) _Float16 v16h;
typedef __attribute__((ext_vector_type(8)))  _Float16 v8h;
typedef __attribute__((ext_vector_type(8)))  float    v8f;
typedef __attribute__((ext_vector_type(4)))  unsigned int u32x4;
typedef __attribute__((ext_vector_type(8)))  int i32x8;
typedef __attribute__((ext_vector_type(4)))  int i32x4;

#if defined(__has_builtin)
#  if __has_builtin(__builtin_amdgcn_tensor_load_to_lds) && __has_builtin(__builtin_amdgcn_s_wait_tensorcnt)
#    define USE_TDM 1
#  endif
#endif
#ifndef USE_TDM
#  define USE_TDM 0
#endif

__device__ __forceinline__ v8f wmma_f16(v16h a, v16h b, v8f c) {
  return __builtin_amdgcn_wmma_f32_16x16x32_f16(false, a, false, b, (short)0, c, false, false);
}

// Load an A/B fragment half-row: elements 0..7 at +8*g, elements 8..15 at +16+8*g
__device__ __forceinline__ v16h load_frag(const _Float16* base, int g) {
  v8h lo = *(const v8h*)(base + 8*g);
  v8h hi = *(const v8h*)(base + 16 + 8*g);
  v16h r;
#pragma unroll
  for (int i = 0; i < 8; ++i) { r[i] = lo[i]; r[i+8] = hi[i]; }
  return r;
}

#if USE_TDM
// 2-D TDM load: tile (tile1 rows x tile0 f16 elements), global row stride = strideElems,
// into LDS at lds_ptr. Descriptor layout per CDNA5 ISA sec 8.3/8.4 (D# groups 0/1).
__device__ __forceinline__ void tdm_load_2d(void* lds_ptr, const void* gptr,
                                            unsigned tile0, unsigned tile1,
                                            unsigned strideElems) {
  u32x4 g0 = {0u, 0u, 0u, 0u};
  i32x8 g1 = {0, 0, 0, 0, 0, 0, 0, 0};
  i32x4 g2 = {0, 0, 0, 0};
  i32x4 g3 = {0, 0, 0, 0};
  unsigned long long ga = (unsigned long long)(size_t)gptr;
  unsigned ldsOff = (unsigned)(size_t)lds_ptr;      // low 32 bits = LDS byte offset
  const unsigned td0 = 1u << 30, td1 = 1u << 30;    // huge tensor dims: tile never OOB
  g0[0] = 1u;                                       // count = 1 valid descriptor
  g0[1] = ldsOff;                                   // lds_addr
  g0[2] = (unsigned)ga;                             // global_addr[31:0]
  g0[3] = (unsigned)(ga >> 32) | (2u << 30);        // global_addr[56:32] | type=2
  g1[0] = (int)(1u << 16);                          // data_size = 1 -> 2 bytes/elem
  g1[1] = (int)((td0 & 0xFFFFu) << 16);             // tensor_dim0[15:0]
  g1[2] = (int)((td0 >> 16) | ((td1 & 0xFFFFu) << 16)); // td0[31:16] | td1[15:0]
  g1[3] = (int)((td1 >> 16) | (tile0 << 16));       // td1[31:16] | tile_dim0
  g1[4] = (int)(tile1 & 0xFFFFu);                   // tile_dim1 (tile_dim2 = 0)
  g1[5] = (int)strideElems;                         // tensor_dim0_stride[31:0]
#if __clang_major__ >= 23
  i32x8 g4 = {0, 0, 0, 0, 0, 0, 0, 0};
  __builtin_amdgcn_tensor_load_to_lds(g0, g1, g2, g3, g4, 0);
#else
  __builtin_amdgcn_tensor_load_to_lds(g0, g1, g2, g3, 0);
#endif
}
#endif

// Fallback cooperative staging: 64x32 f16 panel, 256 threads, one b128 each.
__device__ __forceinline__ void stage_panel(_Float16* dst, const _Float16* src, int K, int tid) {
  int row = tid >> 2, chunk = tid & 3;
  *(v8h*)(dst + row * 32 + chunk * 8) = *(const v8h*)(src + (long)row * K + chunk * 8);
}

// ---------------- weight packing: Wt[l][n][k] = W[l][k][n], f32 -> f16 ----------------
__global__ void k_pack(const float* __restrict__ W, _Float16* __restrict__ out,
                       int K, int N, long lsOut, long offOut) {
  long idx = (long)blockIdx.x * blockDim.x + threadIdx.x;
  long tot = (long)NL * K * N;
  if (idx >= tot) return;
  long per = (long)K * N;
  int l = (int)(idx / per);
  long r = idx % per;
  int n = (int)(r / K), k = (int)(r % K);
  out[(long)l * lsOut + offOut + (long)n * K + k] =
      (_Float16)W[(long)l * per + (long)k * N + n];
}

__global__ void k_pack_bias(const float* __restrict__ bq, const float* __restrict__ bk,
                            const float* __restrict__ bv, float* __restrict__ out) {
  int idx = blockIdx.x * blockDim.x + threadIdx.x;
  if (idx >= NL * QKVN) return;
  int l = idx / QKVN, j = idx % QKVN, part = j / DD, n = j % DD;
  const float* s = (part == 0) ? bq : (part == 1) ? bk : bv;
  out[idx] = s[l * DD + n];
}

// ---------------- RoPE tables ----------------
__global__ void k_rope_tab(float* __restrict__ cosT, float* __restrict__ sinT) {
  int l = blockIdx.x, d = threadIdx.x;       // d in [0,48)
  int j = d % 24;
  float inv = powf(10000.0f, -(float)j / 24.0f);
  int tp = l >> 8, hp = (l >> 4) & 15, wp = l & 15;
  float pos = (j < 6) ? (float)tp : (j < 15) ? (float)hp : (float)wp;
  float f = pos * inv;
  cosT[l * HD + d] = cosf(f);
  sinT[l * HD + d] = sinf(f);
}

// ---------------- unshuffle + input RMSNorm -> x (f32) ----------------
__global__ void k_build_x(const float* __restrict__ vis, const int* __restrict__ ids_restore,
                          const float* __restrict__ mask_token, const float* __restrict__ lnw,
                          float* __restrict__ x) {
  __shared__ float red[128];
  int m = blockIdx.x, t = threadIdx.x;
  int b = m >> 11, l = m & (LL - 1);
  int src = ids_restore[b * LL + l];
  float v[3]; float s = 0.f;
#pragma unroll
  for (int i = 0; i < 3; ++i) {
    int d = t + i * 128;
    float vv = (src < NVIS) ? vis[((long)b * NVIS + src) * DD + d] : mask_token[d];
    v[i] = vv; s += vv * vv;
  }
  red[t] = s; __syncthreads();
  for (int st = 64; st > 0; st >>= 1) { if (t < st) red[t] += red[t + st]; __syncthreads(); }
  float rs = rsqrtf(red[0] / (float)DD + 1e-6f);
#pragma unroll
  for (int i = 0; i < 3; ++i) {
    int d = t + i * 128;
    x[(long)m * DD + d] = v[i] * rs * lnw[d];
  }
}

// ---------------- RMSNorm of residual stream -> f16 ----------------
__global__ void k_rms(const float* __restrict__ x, const float* __restrict__ w,
                      _Float16* __restrict__ out) {
  __shared__ float red[128];
  int m = blockIdx.x, t = threadIdx.x;
  const float* row = x + (long)m * DD;
  float v[3]; float s = 0.f;
#pragma unroll
  for (int i = 0; i < 3; ++i) { v[i] = row[t + i * 128]; s += v[i] * v[i]; }
  red[t] = s; __syncthreads();
  for (int st = 64; st > 0; st >>= 1) { if (t < st) red[t] += red[t + st]; __syncthreads(); }
  float rs = rsqrtf(red[0] / (float)DD + 1e-6f);
#pragma unroll
  for (int i = 0; i < 3; ++i) {
    int d = t + i * 128;
    out[(long)m * DD + d] = (_Float16)(v[i] * rs * w[d]);
  }
}

// ---------------- LDS-staged WMMA GEMM, 64x64 tile per 8-wave block ----------------
// C[m][n] = A[m][:] . Bt[n][:] + bias (+resid)(+gelu); A:[M][K] f16, Bt:[N][K] f16.
__global__ void __launch_bounds__(256) k_gemm(
    const _Float16* __restrict__ A, const _Float16* __restrict__ Bt,
    const float* __restrict__ bias, const float* __restrict__ resid,
    float* __restrict__ Cf, _Float16* __restrict__ Ch,
    int N, int K, int act) {
  __shared__ __align__(16) _Float16 As[2][64 * 32];
  __shared__ __align__(16) _Float16 Bs[2][64 * 32];
  const int tid  = threadIdx.x;
  const int lane = tid & 31, wave = tid >> 5;
  const int li = lane & 15, g = lane >> 4;
  const int msub = wave & 3;      // 0..3 : rows msub*16..+16
  const int npair = wave >> 2;    // 0..1 : cols npair*32..+32
  const int nt = blockIdx.x, mt = blockIdx.y;
  const _Float16* Ag = A  + (long)mt * 64 * K;
  const _Float16* Bg = Bt + (long)nt * 64 * K;

  v8f c0 = {}, c1 = {};

#if USE_TDM
  if (wave == 0) {
    tdm_load_2d(As[0], Ag, 32, 64, (unsigned)K);
    tdm_load_2d(Bs[0], Bg, 32, 64, (unsigned)K);
    __builtin_amdgcn_s_wait_tensorcnt((unsigned short)0);
  }
  __syncthreads();
#else
  stage_panel(As[0], Ag, K, tid);
  stage_panel(Bs[0], Bg, K, tid);
  __syncthreads();
#endif

  int cur = 0;
  for (int kk = 0; kk < K; kk += 32) {
    int nxt = cur ^ 1;
#if USE_TDM
    if ((kk + 32 < K) && wave == 0) {
      tdm_load_2d(As[nxt], Ag + kk + 32, 32, 64, (unsigned)K);
      tdm_load_2d(Bs[nxt], Bg + kk + 32, 32, 64, (unsigned)K);
    }
#endif
    v16h a  = load_frag(As[cur] + (msub * 16 + li) * 32, g);
    v16h b0 = load_frag(Bs[cur] + (npair * 32 + li) * 32, g);
    v16h b1 = load_frag(Bs[cur] + (npair * 32 + 16 + li) * 32, g);
    c0 = wmma_f16(a, b0, c0);
    c1 = wmma_f16(a, b1, c1);
#if USE_TDM
    if ((kk + 32 < K) && wave == 0) __builtin_amdgcn_s_wait_tensorcnt((unsigned short)0);
    __syncthreads();
#else
    __syncthreads();
    if (kk + 32 < K) {
      stage_panel(As[nxt], Ag + kk + 32, K, tid);
      stage_panel(Bs[nxt], Bg + kk + 32, K, tid);
      __syncthreads();
    }
#endif
    cur = nxt;
  }

  const int ncol0 = nt * 64 + npair * 32 + li;
  float bn0 = bias ? bias[ncol0]      : 0.f;
  float bn1 = bias ? bias[ncol0 + 16] : 0.f;
#pragma unroll
  for (int r = 0; r < 8; ++r) {
    int mm = mt * 64 + msub * 16 + r + 8 * g;
    long i0 = (long)mm * N + ncol0;
    long i1 = i0 + 16;
    float v0 = c0[r] + bn0;
    float v1 = c1[r] + bn1;
    if (resid) { v0 += resid[i0]; v1 += resid[i1]; }
    if (act == 1) {
      v0 = 0.5f * v0 * (1.0f + erff(v0 * 0.70710678118654752f));
      v1 = 0.5f * v1 * (1.0f + erff(v1 * 0.70710678118654752f));
    }
    if (Cf) { Cf[i0] = v0; Cf[i1] = v1; }
    if (Ch) { Ch[i0] = (_Float16)v0; Ch[i1] = (_Float16)v1; }
  }
}

// ---------------- RoPE apply: qkv(f32) -> Qh,Kh [bh][l][64] f16 ; Vh [bh][64][l] f16 ----
__global__ void k_rope_apply(const float* __restrict__ qkv, const float* __restrict__ cosT,
                             const float* __restrict__ sinT,
                             _Float16* __restrict__ Qh, _Float16* __restrict__ Kh,
                             _Float16* __restrict__ Vh) {
  int m = blockIdx.x;                 // b*L + l
  int t = threadIdx.x;                // 0..383  (NH * HDP)
  int h = t / HDP, d = t % HDP;
  int b = m >> 11, l = m & (LL - 1);
  long obase = ((long)(b * NH + h) * LL + l) * HDP + d;
  long vbase = ((long)(b * NH + h) * HDP + d) * LL + l;
  if (d >= HD) { Qh[obase] = (_Float16)0.f; Kh[obase] = (_Float16)0.f; Vh[vbase] = (_Float16)0.f; return; }
  const float* row = qkv + (long)m * QKVN;
  float c = cosT[l * HD + d], s = sinT[l * HD + d];
  float q = row[h * HD + d];
  float k = row[DD + h * HD + d];
  float v = row[2 * DD + h * HD + d];
  float qr = (d < 24) ? -row[h * HD + d + 24] : row[h * HD + d - 24];
  float kr = (d < 24) ? -row[DD + h * HD + d + 24] : row[DD + h * HD + d - 24];
  Qh[obase] = (_Float16)(q * c + qr * s);
  Kh[obase] = (_Float16)(k * c + kr * s);
  Vh[vbase] = (_Float16)v;
}

// ---------------- flash attention, one wave per 16-query tile per (b,h) ----------------
__global__ void k_attn(const _Float16* __restrict__ Qh, const _Float16* __restrict__ Kh,
                       const _Float16* __restrict__ Vh, _Float16* __restrict__ AO) {
  __shared__ __align__(16) _Float16 P[16 * 32];
  const int lane = threadIdx.x & 31;
  const int li = lane & 15, g = lane >> 4;
  const int qt = blockIdx.x;          // query tile (16 tokens)
  const int bh = blockIdx.y;          // b*NH + h
  const int b = bh / NH, h = bh % NH;
  const float scale = 0.14433756729740645f;  // 1/sqrt(48)

  const long qbase = ((long)bh * LL + qt * 16 + li) * HDP;
  v16h qa0 = load_frag(Qh + qbase + 0, g);
  v16h qa1 = load_frag(Qh + qbase + 32, g);

  float rowM[8], rowL[8];
  v8f acc[4];
#pragma unroll
  for (int r = 0; r < 8; ++r) { rowM[r] = -1e30f; rowL[r] = 0.f; }
#pragma unroll
  for (int j = 0; j < 4; ++j) acc[j] = (v8f){};

  // frame-causal: keys with frame < query frame are fully masked -> skip
  const int kstart = ((qt * 16) >> 8) << 8;

  for (int kb = kstart; kb < LL; kb += 32) {
    if (kb + 32 < LL) {   // prefetch next K/V tiles -> global_prefetch_b8
      __builtin_prefetch(Kh + ((long)bh * LL + kb + 32 + li) * HDP, 0, 0);
      __builtin_prefetch(Vh + ((long)bh * HDP + li) * LL + kb + 32, 0, 0);
    }
    float p0[8], p1[8], mx[8];
    {
      long kbase = ((long)bh * LL + kb + li) * HDP;
      v16h ka = load_frag(Kh + kbase + 0, g);
      v16h kb2 = load_frag(Kh + kbase + 32, g);
      v8f s = {};
      s = wmma_f16(qa0, ka, s);
      s = wmma_f16(qa1, kb2, s);
#pragma unroll
      for (int r = 0; r < 8; ++r) p0[r] = s[r] * scale;
    }
    {
      long kbase = ((long)bh * LL + kb + 16 + li) * HDP;
      v16h ka = load_frag(Kh + kbase + 0, g);
      v16h kb2 = load_frag(Kh + kbase + 32, g);
      v8f s = {};
      s = wmma_f16(qa0, ka, s);
      s = wmma_f16(qa1, kb2, s);
#pragma unroll
      for (int r = 0; r < 8; ++r) p1[r] = s[r] * scale;
    }
#pragma unroll
    for (int r = 0; r < 8; ++r) mx[r] = fmaxf(p0[r], p1[r]);
#pragma unroll
    for (int msk = 8; msk >= 1; msk >>= 1)
#pragma unroll
      for (int r = 0; r < 8; ++r) mx[r] = fmaxf(mx[r], __shfl_xor(mx[r], msk, 32));

    float sum[8];
#pragma unroll
    for (int r = 0; r < 8; ++r) {
      float mnew = fmaxf(rowM[r], mx[r]);
      float alpha = expf(rowM[r] - mnew);
      p0[r] = expf(p0[r] - mnew);
      p1[r] = expf(p1[r] - mnew);
      sum[r] = p0[r] + p1[r];
      rowL[r] = rowL[r] * alpha;
      rowM[r] = mnew;
#pragma unroll
      for (int j = 0; j < 4; ++j) acc[j][r] *= alpha;
    }
#pragma unroll
    for (int msk = 8; msk >= 1; msk >>= 1)
#pragma unroll
      for (int r = 0; r < 8; ++r) sum[r] += __shfl_xor(sum[r], msk, 32);
#pragma unroll
    for (int r = 0; r < 8; ++r) rowL[r] += sum[r];

    // stage P (16x32) to LDS in row-major, re-load as A fragment
#pragma unroll
    for (int r = 0; r < 8; ++r) {
      P[(r + 8 * g) * 32 + li]      = (_Float16)p0[r];
      P[(r + 8 * g) * 32 + 16 + li] = (_Float16)p1[r];
    }
    __syncthreads();
    v16h pa = load_frag(P + li * 32, g);
#pragma unroll
    for (int j = 0; j < 4; ++j) {
      long vb = ((long)bh * HDP + j * 16 + li) * LL + kb;
      v16h vf = load_frag(Vh + vb, g);
      acc[j] = wmma_f16(pa, vf, acc[j]);
    }
    __syncthreads();
  }

#pragma unroll
  for (int r = 0; r < 8; ++r) {
    float inv = 1.0f / rowL[r];
    long mg = (long)b * LL + qt * 16 + r + 8 * g;
#pragma unroll
    for (int j = 0; j < 3; ++j) {   // cols 48..63 are padding, dropped
      int n = j * 16 + li;
      AO[mg * DD + h * HD + n] = (_Float16)(acc[j][r] * inv);
    }
  }
}

// ---------------- output: copy + stable mask gather ----------------
__global__ void k_copy(const float* __restrict__ src, float* __restrict__ dst, long n) {
  long i = (long)blockIdx.x * blockDim.x + threadIdx.x;
  if (i < n) dst[i] = src[i];
}

__global__ void k_ranks(const int* __restrict__ mask, int* __restrict__ ranks) {
  int idx = blockIdx.x * blockDim.x + threadIdx.x;
  if (idx >= MTOT) return;
  int b = idx >> 11, l = idx & (LL - 1);
  const int* mr = mask + b * LL;
  int mv = mr[l], cnt = 0;
  for (int i = 0; i < l; ++i) cnt += (mr[i] == mv) ? 1 : 0;
  ranks[idx] = cnt;
}

__global__ void k_gather(const float* __restrict__ x, const int* __restrict__ mask,
                         const int* __restrict__ ranks, float* __restrict__ out) {
  const long FULLN = (long)BB * LL * DD;
  const long VISN  = (long)BB * NVIS * DD;
  int tok = blockIdx.x;
  int b = tok >> 11, l = tok & (LL - 1);
  int mv = mask[b * LL + l];
  int r = ranks[tok];
  float* dst = mv ? out + FULLN + ((long)b * NVIS + r) * DD
                  : out + FULLN + VISN + ((long)b * (LL - NVIS) + r) * DD;
  const float* src = x + (long)tok * DD;
  for (int d = threadIdx.x; d < DD; d += blockDim.x) dst[d] = src[d];
}

// ---------------- host launcher ----------------
static inline char* wsalloc(char* base, size_t& off, size_t bytes) {
  char* p = base + off;
  off += (bytes + 255) & ~(size_t)255;
  return p;
}

extern "C" void kernel_launch(void* const* d_in, const int* in_sizes, int n_in,
                              void* d_out, int out_size, void* d_ws, size_t ws_size,
                              hipStream_t stream) {
  (void)in_sizes; (void)n_in; (void)out_size; (void)ws_size;
  const float* vis        = (const float*)d_in[0];
  const int*   ids        = (const int*)d_in[1];
  const int*   mask       = (const int*)d_in[2];
  const float* mask_token = (const float*)d_in[6];
  const float* ln_in_w    = (const float*)d_in[7];
  const float* ln1_w      = (const float*)d_in[8];
  const float* ln2_w      = (const float*)d_in[9];
  const float* Wq = (const float*)d_in[10]; const float* bq = (const float*)d_in[11];
  const float* Wk = (const float*)d_in[12]; const float* bk = (const float*)d_in[13];
  const float* Wv = (const float*)d_in[14]; const float* bv = (const float*)d_in[15];
  const float* Wo = (const float*)d_in[16]; const float* bo = (const float*)d_in[17];
  const float* W1 = (const float*)d_in[18]; const float* b1 = (const float*)d_in[19];
  const float* W2 = (const float*)d_in[20]; const float* b2 = (const float*)d_in[21];
  float* out = (float*)d_out;

  char* base = (char*)d_ws;
  size_t off = 0;
  _Float16* WtQKV = (_Float16*)wsalloc(base, off, (size_t)NL*QKVN*DD * 2);
  _Float16* WtO   = (_Float16*)wsalloc(base, off, (size_t)NL*DD*DD   * 2);
  _Float16* Wt1   = (_Float16*)wsalloc(base, off, (size_t)NL*FF*DD   * 2);
  _Float16* Wt2   = (_Float16*)wsalloc(base, off, (size_t)NL*DD*FF   * 2);
  float*    bqkv  = (float*)   wsalloc(base, off, (size_t)NL*QKVN    * 4);
  float*    x     = (float*)   wsalloc(base, off, (size_t)MTOT*DD    * 4);
  _Float16* hdn   = (_Float16*)wsalloc(base, off, (size_t)MTOT*DD    * 2);
  float*    qkv   = (float*)   wsalloc(base, off, (size_t)MTOT*QKVN  * 4);
  _Float16* Qh    = (_Float16*)wsalloc(base, off, (size_t)BB*NH*LL*HDP * 2);
  _Float16* Kh    = (_Float16*)wsalloc(base, off, (size_t)BB*NH*LL*HDP * 2);
  _Float16* Vh    = (_Float16*)wsalloc(base, off, (size_t)BB*NH*HDP*LL * 2);
  _Float16* AO    = (_Float16*)wsalloc(base, off, (size_t)MTOT*DD    * 2);
  _Float16* MH    = (_Float16*)wsalloc(base, off, (size_t)MTOT*FF    * 2);
  float*    cosT  = (float*)   wsalloc(base, off, (size_t)LL*HD      * 4);
  float*    sinT  = (float*)   wsalloc(base, off, (size_t)LL*HD      * 4);
  int*      ranks = (int*)     wsalloc(base, off, (size_t)MTOT       * 4);

  // ---- weight packing ----
  {
    long n1 = (long)NL * DD * DD;   int g1 = (int)((n1 + 255) / 256);
    k_pack<<<g1, 256, 0, stream>>>(Wq, WtQKV, DD, DD, (long)QKVN*DD, 0L);
    k_pack<<<g1, 256, 0, stream>>>(Wk, WtQKV, DD, DD, (long)QKVN*DD, (long)DD*DD);
    k_pack<<<g1, 256, 0, stream>>>(Wv, WtQKV, DD, DD, (long)QKVN*DD, 2L*DD*DD);
    k_pack<<<g1, 256, 0, stream>>>(Wo, WtO,   DD, DD, (long)DD*DD,   0L);
    long n2 = (long)NL * DD * FF;   int g2 = (int)((n2 + 255) / 256);
    k_pack<<<g2, 256, 0, stream>>>(W1, Wt1, DD, FF, (long)FF*DD, 0L);
    k_pack<<<g2, 256, 0, stream>>>(W2, Wt2, FF, DD, (long)DD*FF, 0L);
    k_pack_bias<<<(NL*QKVN + 255)/256, 256, 0, stream>>>(bq, bk, bv, bqkv);
  }

  k_rope_tab<<<LL, HD, 0, stream>>>(cosT, sinT);
  k_build_x<<<MTOT, 128, 0, stream>>>(vis, ids, mask_token, ln_in_w, x);

  for (int l = 0; l < NL; ++l) {
    const _Float16* wqkv_l = WtQKV + (long)l * QKVN * DD;
    const _Float16* wo_l   = WtO   + (long)l * DD * DD;
    const _Float16* w1_l   = Wt1   + (long)l * FF * DD;
    const _Float16* w2_l   = Wt2   + (long)l * DD * FF;

    // hdn = rms(x, ln1)
    k_rms<<<MTOT, 128, 0, stream>>>(x, ln1_w + (long)l * DD, hdn);
    // qkv = hdn @ Wqkv + bqkv   (N=1152, K=384)
    k_gemm<<<dim3(QKVN/64, MTOT/64), 256, 0, stream>>>(
        hdn, wqkv_l, bqkv + (long)l * QKVN, nullptr, qkv, nullptr, QKVN, DD, 0);
    // RoPE + layout for attention
    k_rope_apply<<<MTOT, NH*HDP, 0, stream>>>(qkv, cosT, sinT, Qh, Kh, Vh);
    // flash attention
    k_attn<<<dim3(LL/16, BB*NH), 32, 0, stream>>>(Qh, Kh, Vh, AO);
    // x += AO @ Wo + bo
    k_gemm<<<dim3(DD/64, MTOT/64), 256, 0, stream>>>(
        AO, wo_l, bo + (long)l * DD, x, x, nullptr, DD, DD, 0);
    // hdn = rms(x, ln2)
    k_rms<<<MTOT, 128, 0, stream>>>(x, ln2_w + (long)l * DD, hdn);
    // MH = gelu(hdn @ W1 + b1)   (f16 out)
    k_gemm<<<dim3(FF/64, MTOT/64), 256, 0, stream>>>(
        hdn, w1_l, b1 + (long)l * FF, nullptr, nullptr, MH, FF, DD, 1);
    // x += MH @ W2 + b2
    k_gemm<<<dim3(DD/64, MTOT/64), 256, 0, stream>>>(
        MH, w2_l, b2 + (long)l * DD, x, x, nullptr, DD, FF, 0);
  }

  // outputs: full, visible, masked
  long fulln = (long)BB * LL * DD;
  k_copy<<<(int)((fulln + 255)/256), 256, 0, stream>>>(x, out, fulln);
  k_ranks<<<(MTOT + 255)/256, 256, 0, stream>>>(mask, ranks);
  k_gather<<<MTOT, 128, 0, stream>>>(x, mask, ranks, out);
}
